// CrossWinAttention_29566554866343
// MI455X (gfx1250) — matrix operand order repaired and checked by reading, not compiled, not run on hardware
//
#include <hip/hip_runtime.h>

typedef __attribute__((ext_vector_type(16))) _Float16 v16h;
typedef __attribute__((ext_vector_type(8)))  _Float16 v8h;
typedef __attribute__((ext_vector_type(8)))  float    v8f;

#define NB 2
#define NA 9      // agents
#define NX 12
#define NY 12
#define D  128
#define T  144    // tokens per window = NA*16
#define H  4
#define DH 32
#define SCALE 0.17677669529663687f   // 1/sqrt(32)

// ---- LDS carve (bytes). Total 318464 <= 320KB WGP LDS ----
#define KF_OFF   0         // Kf   : 144x128 f16 (36864)
#define QF_OFF   36864     // Qf   : 144x128 f16 (scale folded in)
#define VT_OFF   73728     // Vt   : 128x160 f16, transposed [d][tok], tok padded->160
#define OB_OFF   114688    // O    : 144x128 f16
#define WB_OFF   151552    // Wbuf : 128x128 f16, transposed [n][k]
#define S_OFF    184320    // S    : 144x144 f32 ; overlay: Xln 144x128 f16 during proj
#define P_OFF    267264    // P    : 144x160 f16, k padded->160 (zeros)
#define PAR_OFF  313344    // params: 10 x 128 f32
#define SMEM_BYTES 318464

__device__ __forceinline__ v16h ld16(const _Float16* p0, const _Float16* p1) {
  v8h lo = *(const v8h*)p0;
  v8h hi = *(const v8h*)p1;
  return __builtin_shufflevector(lo, hi, 0,1,2,3,4,5,6,7,8,9,10,11,12,13,14,15);
}

__device__ __forceinline__ v8f wmma_f16(v16h a, v16h b, v8f c) {
  // emits v_wmma_f32_16x16x32_f16
  return __builtin_amdgcn_wmma_f32_16x16x32_f16(false, a, false, b, (short)0, c, false, false);
}

__device__ __forceinline__ float wsum(float v) {
#pragma unroll
  for (int off = 16; off >= 1; off >>= 1) v += __shfl_xor(v, off, 32);
  return v;
}
__device__ __forceinline__ float wmax(float v) {
#pragma unroll
  for (int off = 16; off >= 1; off >>= 1) v = fmaxf(v, __shfl_xor(v, off, 32));
  return v;
}

// LayerNorm + (X @ W + bias)*scale, result into Qf(mode 0)/Kf(mode 1)/Vt-transposed(mode 2)
__device__ void ln_project(const float* __restrict__ src, const float* __restrict__ Wsrc,
                           const float* lnw, const float* lnb, const float* bias,
                           float scale, int mode, char* smem, int bb, int xx, int yy) {
  const int tid = threadIdx.x, lane = tid & 31, wave = tid >> 5, g = lane >> 4;
  _Float16* Xln  = (_Float16*)(smem + S_OFF);
  _Float16* Wbuf = (_Float16*)(smem + WB_OFF);
  _Float16* Qf   = (_Float16*)(smem + QF_OFF);
  _Float16* Kf   = (_Float16*)(smem + KF_OFF);
  _Float16* Vt   = (_Float16*)(smem + VT_OFF);

  // stage weight transposed: Wbuf[n][k] = W[k][n], f32->f16 (coalesced global reads)
  for (int i = tid; i < D * D; i += 256) {
    int kk = i >> 7, nn = i & 127;
    Wbuf[nn * D + kk] = (_Float16)Wsrc[i];
  }
  // LayerNorm each token (one wave per token, 4 f32 per lane)
  for (int t = wave; t < T; t += 8) {
    int agent = t >> 4, w1 = (t >> 2) & 3, w2 = t & 3;
    long base = ((((((long)bb * NA + agent) * NX + xx) * NY + yy) * 4 + w1) * 4 + w2) * (long)D;
    float4 v = *(const float4*)(src + base + lane * 4);
    float m  = wsum(v.x + v.y + v.z + v.w) * (1.f / 128.f);
    float dx = v.x - m, dy = v.y - m, dz = v.z - m, dw = v.w - m;
    float inv = rsqrtf(wsum(dx*dx + dy*dy + dz*dz + dw*dw) * (1.f / 128.f) + 1e-5f);
    int d0 = lane * 4;
    Xln[t * D + d0 + 0] = (_Float16)(dx * inv * lnw[d0 + 0] + lnb[d0 + 0]);
    Xln[t * D + d0 + 1] = (_Float16)(dy * inv * lnw[d0 + 1] + lnb[d0 + 1]);
    Xln[t * D + d0 + 2] = (_Float16)(dz * inv * lnw[d0 + 2] + lnb[d0 + 2]);
    Xln[t * D + d0 + 3] = (_Float16)(dw * inv * lnw[d0 + 3] + lnb[d0 + 3]);
  }
  __syncthreads();
  // GEMM: X(144x128) @ W(128x128); 9x8 = 72 output tiles over 8 waves
  for (int tile = wave; tile < 72; tile += 8) {
    int tm = tile >> 3, tn = tile & 7;
    int mrow = tm * 16 + (lane & 15);
    int ncol = tn * 16 + (lane & 15);
    v8f c = {};
#pragma unroll
    for (int kb = 0; kb < 128; kb += 32) {
      v16h a = ld16(&Xln[mrow * D + kb + g * 16], &Xln[mrow * D + kb + g * 16 + 8]);
      const _Float16* wc = &Wbuf[ncol * D + kb + g * 16];
      v16h bf = ld16(wc, wc + 8);
      c = wmma_f16(a, bf, c);
    }
    float bv = bias[ncol];
#pragma unroll
    for (int r = 0; r < 8; ++r) {
      int row = tm * 16 + r + g * 8;
      float val = (c[r] + bv) * scale;
      if (mode == 2)      Vt[ncol * 160 + row] = (_Float16)val;   // transposed for PV B-frags
      else if (mode == 1) Kf[row * D + ncol]   = (_Float16)val;
      else                Qf[row * D + ncol]   = (_Float16)val;
    }
  }
  __syncthreads();
}

extern "C" __global__ void __launch_bounds__(256, 1)
cwa_fused_kernel(const float* __restrict__ q, const float* __restrict__ k,
                 const float* __restrict__ v, const float* __restrict__ skip,
                 const float* __restrict__ ln_q_w, const float* __restrict__ ln_q_b,
                 const float* __restrict__ ln_k_w, const float* __restrict__ ln_k_b,
                 const float* __restrict__ ln_v_w, const float* __restrict__ ln_v_b,
                 const float* __restrict__ Wq, const float* __restrict__ bq,
                 const float* __restrict__ Wk, const float* __restrict__ bk,
                 const float* __restrict__ Wv, const float* __restrict__ bv,
                 const float* __restrict__ Wp, const float* __restrict__ bp,
                 float* __restrict__ out) {
  extern __shared__ char smem[];
  const int tid = threadIdx.x, lane = tid & 31, wave = tid >> 5, g = lane >> 4;
  const int blk = blockIdx.x;
  const int bb = blk / 144, wxy = blk % 144, xx = wxy / 12, yy = wxy % 12;

  float*    par = (float*)(smem + PAR_OFF);
  _Float16* Vt  = (_Float16*)(smem + VT_OFF);
  _Float16* P   = (_Float16*)(smem + P_OFF);
  _Float16* Qf  = (_Float16*)(smem + QF_OFF);
  _Float16* Kf  = (_Float16*)(smem + KF_OFF);
  _Float16* Ob  = (_Float16*)(smem + OB_OFF);
  _Float16* Wb  = (_Float16*)(smem + WB_OFF);
  float*    S   = (float*)(smem + S_OFF);

  // params -> LDS; zero the K-dim padding of Vt and P once
  if (tid < 128) {
    par[tid]        = ln_q_w[tid]; par[128 + tid]  = ln_q_b[tid];
    par[256 + tid]  = ln_k_w[tid]; par[384 + tid]  = ln_k_b[tid];
    par[512 + tid]  = ln_v_w[tid]; par[640 + tid]  = ln_v_b[tid];
    par[768 + tid]  = bq[tid];     par[896 + tid]  = bk[tid];
    par[1024 + tid] = bv[tid];     par[1152 + tid] = bp[tid];
  }
  for (int i = tid; i < 128 * 16; i += 256) Vt[(i >> 4) * 160 + 144 + (i & 15)] = (_Float16)0.f;
  for (int i = tid; i < 144 * 16; i += 256) P [(i >> 4) * 160 + 144 + (i & 15)] = (_Float16)0.f;
  __syncthreads();

  // ---- QKV projections (LN + GEMM + bias), all WMMA ----
  ln_project(q, Wq, par + 0,   par + 128, par + 768,  SCALE, 0, smem, bb, xx, yy);
  ln_project(k, Wk, par + 256, par + 384, par + 896,  1.f,   1, smem, bb, xx, yy);
  ln_project(v, Wv, par + 512, par + 640, par + 1024, 1.f,   2, smem, bb, xx, yy);

  // ---- attention, head by head (S buffer shared) ----
  for (int h = 0; h < H; ++h) {
    const int hb = h * DH;
    // S = Q_h @ K_h^T  (scale already folded into Qf); K=32 -> one WMMA per 16x16 tile
    for (int t = wave; t < 81; t += 8) {
      int tm = t / 9, tn = t % 9;
      int mrow = tm * 16 + (lane & 15);
      int kc   = tn * 16 + (lane & 15);
      v16h a = ld16(&Qf[mrow * D + hb + g * 16], &Qf[mrow * D + hb + g * 16 + 8]);
      const _Float16* kp = &Kf[kc * D + hb + g * 16];
      v16h bf = ld16(kp, kp + 8);
      v8f c = {};
      c = wmma_f16(a, bf, c);
#pragma unroll
      for (int r = 0; r < 8; ++r) S[(tm * 16 + r + g * 8) * T + kc] = c[r];
    }
    __syncthreads();
    // softmax rows -> P (f16, width padded to 160 with zeros)
    for (int row = wave; row < T; row += 8) {
      const float* sr = &S[row * T];
      float x0 = sr[lane], x1 = sr[lane + 32], x2 = sr[lane + 64], x3 = sr[lane + 96];
      float x4 = (lane < 16) ? sr[lane + 128] : -3.0e38f;
      float mx = wmax(fmaxf(fmaxf(x0, x1), fmaxf(fmaxf(x2, x3), x4)));
      float e0 = __expf(x0 - mx), e1 = __expf(x1 - mx);
      float e2 = __expf(x2 - mx), e3 = __expf(x3 - mx);
      float e4 = (lane < 16) ? __expf(x4 - mx) : 0.f;
      float inv = 1.f / wsum(e0 + e1 + e2 + e3 + e4);
      _Float16* pr = &P[row * 160];
      pr[lane]      = (_Float16)(e0 * inv);
      pr[lane + 32] = (_Float16)(e1 * inv);
      pr[lane + 64] = (_Float16)(e2 * inv);
      pr[lane + 96] = (_Float16)(e3 * inv);
      if (lane < 16) pr[lane + 128] = (_Float16)(e4 * inv);
    }
    __syncthreads();
    // O_h = P @ V_h  (Vt transposed -> contiguous B-frags; padded K, zeros contribute 0)
    for (int t = wave; t < 18; t += 8) {
      int tm = t >> 1, tn = t & 1;
      int mrow = tm * 16 + (lane & 15);
      int vrow = hb + tn * 16 + (lane & 15);
      v8f c = {};
#pragma unroll
      for (int kb = 0; kb < 160; kb += 32) {
        v16h a = ld16(&P[mrow * 160 + kb + g * 16], &P[mrow * 160 + kb + g * 16 + 8]);
        const _Float16* vp = &Vt[vrow * 160 + kb + g * 16];
        v16h bf = ld16(vp, vp + 8);
        c = wmma_f16(a, bf, c);
      }
#pragma unroll
      for (int r = 0; r < 8; ++r)
        Ob[(tm * 16 + r + g * 8) * D + hb + tn * 16 + (lane & 15)] = (_Float16)c[r];
    }
    __syncthreads();
  }

  // ---- Z = O @ Wp + bp, mean over agents, + skip ----
  // Each 16-row M-tile is exactly one agent, so accumulating all 9 M-tiles in one
  // WMMA accumulator yields the agent-sum directly. One wave per N-tile (8 N-tiles x 9 M-tiles = 72 tiles).
  for (int i = tid; i < D * D; i += 256) {
    int kk = i >> 7, nn = i & 127;
    Wb[nn * D + kk] = (_Float16)Wp[i];
  }
  __syncthreads();
  {
    int tn = wave;                       // 8 waves <-> 8 column tiles
    int ncol = tn * 16 + (lane & 15);
    v8f c = {};
    for (int tm = 0; tm < 9; ++tm) {     // sum over agents in the accumulator
      int mrow = tm * 16 + (lane & 15);
#pragma unroll
      for (int kb = 0; kb < 128; kb += 32) {
        v16h a = ld16(&Ob[mrow * D + kb + g * 16], &Ob[mrow * D + kb + g * 16 + 8]);
        const _Float16* wc = &Wb[ncol * D + kb + g * 16];
        v16h bf = ld16(wc, wc + 8);
        c = wmma_f16(a, bf, c);
      }
    }
    float bpv = par[1152 + ncol];
#pragma unroll
    for (int r = 0; r < 8; ++r) {
      int w = r + g * 8;                 // window position within tile == W1*W2 index
      long o = ((((long)bb * NX + xx) * NY + yy) * 16 + w) * (long)D + ncol;
      out[o] = c[r] * (1.f / 9.f) + bpv + skip[o];
    }
  }
}

extern "C" void kernel_launch(void* const* d_in, const int* in_sizes, int n_in,
                              void* d_out, int out_size, void* d_ws, size_t ws_size,
                              hipStream_t stream) {
  (void)in_sizes; (void)n_in; (void)out_size; (void)d_ws; (void)ws_size;
  const float* q      = (const float*)d_in[0];
  const float* k      = (const float*)d_in[1];
  const float* v      = (const float*)d_in[2];
  const float* skip   = (const float*)d_in[3];
  const float* ln_q_w = (const float*)d_in[4];
  const float* ln_q_b = (const float*)d_in[5];
  const float* ln_k_w = (const float*)d_in[6];
  const float* ln_k_b = (const float*)d_in[7];
  const float* ln_v_w = (const float*)d_in[8];
  const float* ln_v_b = (const float*)d_in[9];
  const float* Wq     = (const float*)d_in[10];
  const float* bq     = (const float*)d_in[11];
  const float* Wk     = (const float*)d_in[12];
  const float* bk     = (const float*)d_in[13];
  const float* Wv     = (const float*)d_in[14];
  const float* bv     = (const float*)d_in[15];
  const float* Wp     = (const float*)d_in[16];
  const float* bp     = (const float*)d_in[17];
  float* out = (float*)d_out;

  dim3 grid(NB * NX * NY);   // 288 workgroups: one per (batch, window)
  dim3 block(256);           // 8 wave32
  hipLaunchKernelGGL(cwa_fused_kernel, grid, block, SMEM_BYTES, stream,
                     q, k, v, skip, ln_q_w, ln_q_b, ln_k_w, ln_k_b, ln_v_w, ln_v_b,
                     Wq, bq, Wk, bk, Wv, bv, Wp, bp, out);
}
